// CausalAttention_20126216750004
// MI455X (gfx1250) — matrix operand order repaired and checked by reading, not compiled, not run on hardware
//
#include <hip/hip_runtime.h>
#include <hip/hip_bf16.h>
#include <math.h>

// ---------------------------------------------------------------------------
// Causal attention fwd for MI455X (gfx1250, wave32, WMMA, LDS-staged GEMMs).
// All GEMMs use v_wmma_f32_16x16x32_bf16 (f32 accumulation). Block-shared
// operands are double-buffered through LDS (one barrier per 32-K chunk).
// ws layout (bytes):
//   [0,   16Mi)  Qb  bf16 [4][2048][1024]   row-major (WMMA A operand)
//   [16Mi,32Mi)  Kb  bf16 [4][2048][1024]   row-major (WMMA B operand of Q.K^T)
//   [32Mi,48Mi)  Vt  bf16 [4][1024][2048]   V transposed (B operand of P.V)
//   [48Mi,54Mi)  Wt  bf16 [3][1024][1024]   W transposed (B operand of X.W)
//   [56Mi,120Mi) S   f32  [4][2048][2048]   masked scores
//     [56Mi,72Mi)  Xb bf16 [8192][1024]     x in bf16 (dead before S written)
//   [120Mi,152Mi)P   bf16 [4][2048][2048]   softmax probabilities
// ---------------------------------------------------------------------------

typedef __bf16 bf16;
typedef __attribute__((ext_vector_type(16))) __bf16 v16bf;
typedef __attribute__((ext_vector_type(8)))  __bf16 v8bf;
typedef __attribute__((ext_vector_type(4)))  __bf16 v4bf;
typedef __attribute__((ext_vector_type(8)))  float  v8f;
typedef __attribute__((ext_vector_type(4)))  float  v4f;

static constexpr int kD  = 1024;  // model dim (D_IN == D_OUT)
static constexpr int kS  = 2048;  // sequence length
static constexpr int kLP = 40;    // LDS row pitch (elems): 20 dwords -> no bank dup

static __device__ __forceinline__ bf16 f2bf(float x) { return (bf16)x; }

static __device__ __forceinline__ v8f wmma_bf16(v16bf a, v16bf b, v8f c) {
  // D = A(16x32 bf16) x B(32x16 bf16) + C(16x16 f32)
  return __builtin_amdgcn_wmma_f32_16x16x32_bf16(
      /*neg_a=*/false, a, /*neg_b=*/false, b,
      /*c_mod=*/(short)0, c, /*reuse_a=*/false, /*reuse_b=*/false);
}

// Build a 16-elem fragment from two 16B halves (works for global or LDS).
static __device__ __forceinline__ v16bf frag2(const bf16* p0, const bf16* p1) {
  const v8bf lo = *(const v8bf*)p0;
  const v8bf hi = *(const v8bf*)p1;
  v16bf r;
#pragma unroll
  for (int e = 0; e < 8; ++e) { r[e] = lo[e]; r[e + 8] = hi[e]; }
  return r;
}

// A fragment from a row-major row: elems 0-15 -> K=h*16+0..15
static __device__ __forceinline__ v16bf load_a16(const bf16* row, int kb, int h) {
  return frag2(row + kb + h * 16, row + kb + h * 16 + 8);
}

// B fragment from global: lane = column, elems 0-15 -> K=h*16+0..15 (32B load)
static __device__ __forceinline__ v16bf load_b16(const bf16* col, int kb, int h) {
  return *(const v16bf*)(col + kb + h * 16);
}

// B fragment from an LDS-staged 32-K column chunk (pitch kLP)
static __device__ __forceinline__ v16bf load_b_lds(const bf16* colbase, int h) {
  return frag2(colbase + h * 16, colbase + h * 16 + 8);
}

// A fragment from an LDS-staged 32-K row chunk (pitch kLP)
static __device__ __forceinline__ v16bf load_a_lds(const bf16* rowbase, int h) {
  return frag2(rowbase + h * 16, rowbase + h * 16 + 8);
}

// ---------------------------------------------------------------------------
// 0) x f32 -> bf16 (4 elements / thread)
__global__ void xconv_k(const float* __restrict__ X, bf16* __restrict__ Xb) {
  const int i4 = (blockIdx.x * 256 + threadIdx.x) * 4;
  const v4f xv = *(const v4f*)(X + i4);
  v4bf o;
#pragma unroll
  for (int e = 0; e < 4; ++e) o[e] = f2bf(xv[e]);
  *(v4bf*)(Xb + i4) = o;
}

// ---------------------------------------------------------------------------
// 1) W [1024][1024] f32 (KxN) -> Wt [1024][1024] bf16 (NxK)
__global__ void wtrans_k(const float* __restrict__ Wq, const float* __restrict__ Wk,
                         const float* __restrict__ Wv, bf16* __restrict__ Wt) {
  const int idx = blockIdx.x * 256 + threadIdx.x;     // 0 .. 1M-1
  const int mat = blockIdx.y;
  const float* W = (mat == 0) ? Wq : ((mat == 1) ? Wk : Wv);
  const int k = idx >> 10, n = idx & 1023;
  Wt[(size_t)mat * kD * kD + (size_t)n * kD + k] = f2bf(W[idx]);
}

// ---------------------------------------------------------------------------
// 2) QKV projection: C = Xb * W. One wave -> 32x64 tile; the 64x32 W tile is
//    shared by all 8 waves and double-buffered through LDS.
__global__ void __launch_bounds__(256) qkv_k(
    const bf16* __restrict__ Xb, const bf16* __restrict__ Wt,
    bf16* __restrict__ Qb, bf16* __restrict__ Kb, bf16* __restrict__ Vt) {
  __shared__ bf16 bufB[2][64 * kLP];
  const int tid = threadIdx.x;
  const int lane = tid & 31, w = tid >> 5;
  const int r = lane & 15, h = lane >> 4;
  const int mt    = blockIdx.x * 8 + w;   // 0..255 (32-row tiles over b*s)
  const int nbase = blockIdx.y * 64;
  const int mat   = blockIdx.z;

  const bf16* xrow0 = Xb + (size_t)(mt * 32 + r) * kD;
  const bf16* xrow1 = xrow0 + (size_t)16 * kD;

  // staging map: thread -> (col 0..63, kk 0,8,16,24), 8 elems (16B) each
  const int fcol = tid >> 2, fkk = (tid & 3) * 8;
  const bf16* wsrc = Wt + (size_t)mat * kD * kD + (size_t)(nbase + fcol) * kD + fkk;
  const int  foff = fcol * kLP + fkk;

  v8bf stg = *(const v8bf*)(wsrc);        // chunk kb=0
  *(v8bf*)(&bufB[0][foff]) = stg;
  __syncthreads();

  const v8f zero = {0.f, 0.f, 0.f, 0.f, 0.f, 0.f, 0.f, 0.f};
  v8f acc[2][4] = {{zero, zero, zero, zero}, {zero, zero, zero, zero}};

  int ib = 0;
  for (int kb = 0; kb < kD; kb += 32) {
    const bool more = (kb + 32) < kD;
    if (more) stg = *(const v8bf*)(wsrc + kb + 32);

    const v16bf a0 = load_a16(xrow0, kb, h);
    const v16bf a1 = load_a16(xrow1, kb, h);
#pragma unroll
    for (int t = 0; t < 4; ++t) {
      const v16bf b = load_b_lds(&bufB[ib][(t * 16 + r) * kLP], h);
      acc[0][t] = wmma_bf16(a0, b, acc[0][t]);
      acc[1][t] = wmma_bf16(a1, b, acc[1][t]);
    }
    if (more) *(v8bf*)(&bufB[ib ^ 1][foff]) = stg;
    __syncthreads();
    ib ^= 1;
  }

#pragma unroll
  for (int u = 0; u < 2; ++u) {
#pragma unroll
    for (int t = 0; t < 4; ++t) {
      const int n = nbase + t * 16 + r;
#pragma unroll
      for (int v = 0; v < 8; ++v) {
        const int gm = mt * 32 + u * 16 + v + 8 * h;   // global row in [0,8192)
        const bf16 val = f2bf(acc[u][t][v]);
        if (mat == 0) {
          Qb[(size_t)gm * kD + n] = val;
        } else if (mat == 1) {
          Kb[(size_t)gm * kD + n] = val;
        } else {
          const int bb = gm >> 11, s = gm & (kS - 1);
          Vt[(size_t)bb * kD * kS + (size_t)n * kS + s] = val;
        }
      }
    }
  }
}

// ---------------------------------------------------------------------------
// 3) S = (Q K^T)/32 with causal mask. One wave -> 32x32 tile; the 32x32 Q
//    tile (same qt for all 8 waves) is staged through LDS. Above-diagonal
//    waves stay in the barrier protocol; fully-dead blocks exit uniformly.
__global__ void __launch_bounds__(256) scores_k(
    const bf16* __restrict__ Qb, const bf16* __restrict__ Kb,
    float* __restrict__ S) {
  const int qt = blockIdx.x;              // 0..63 (32-row q tiles)
  if ((int)blockIdx.y * 8 > qt) return;   // block entirely above the diagonal
  __shared__ bf16 bufA[2][32 * kLP];

  const int tid = threadIdx.x;
  const int lane = tid & 31, w = tid >> 5;
  const int r = lane & 15, h = lane >> 4;
  const int kt = blockIdx.y * 8 + w;      // 0..63 (32-col k tiles)
  const int bz = blockIdx.z;
  const bool active = (kt <= qt);         // wave-uniform

  // staging map: thread -> (row 0..31, kk 0,4,..,28), 4 elems (8B) each
  const int frow = tid >> 3, fkk = (tid & 7) * 4;
  const bf16* qsrc = Qb + ((size_t)bz * kS + qt * 32 + frow) * kD + fkk;
  const int  foff = frow * kLP + fkk;

  v4bf stg = *(const v4bf*)(qsrc);        // chunk eb=0
  *(v4bf*)(&bufA[0][foff]) = stg;
  __syncthreads();

  const bf16* krow0 = Kb + ((size_t)bz * kS + kt * 32 + r) * kD;
  const bf16* krow1 = krow0 + (size_t)16 * kD;

  const v8f zero = {0.f, 0.f, 0.f, 0.f, 0.f, 0.f, 0.f, 0.f};
  v8f acc[2][2] = {{zero, zero}, {zero, zero}};

  int ib = 0;
  for (int eb = 0; eb < kD; eb += 32) {
    const bool more = (eb + 32) < kD;
    if (more) stg = *(const v4bf*)(qsrc + eb + 32);

    if (active) {
      const v16bf a0 = load_a_lds(&bufA[ib][r * kLP], h);
      const v16bf a1 = load_a_lds(&bufA[ib][(16 + r) * kLP], h);
      const v16bf b0 = load_b16(krow0, eb, h);
      const v16bf b1 = load_b16(krow1, eb, h);
      acc[0][0] = wmma_bf16(a0, b0, acc[0][0]);
      acc[0][1] = wmma_bf16(a0, b1, acc[0][1]);
      acc[1][0] = wmma_bf16(a1, b0, acc[1][0]);
      acc[1][1] = wmma_bf16(a1, b1, acc[1][1]);
    }
    if (more) *(v4bf*)(&bufA[ib ^ 1][foff]) = stg;
    __syncthreads();
    ib ^= 1;
  }

  if (active) {
    const float scale = 0.03125f;         // 1/sqrt(1024)
#pragma unroll
    for (int u = 0; u < 2; ++u) {
#pragma unroll
      for (int t = 0; t < 2; ++t) {
#pragma unroll
        for (int v = 0; v < 8; ++v) {
          const int q  = qt * 32 + u * 16 + v + 8 * h;
          const int kc = kt * 32 + t * 16 + r;
          S[((size_t)bz * kS + q) * kS + kc] =
              (kc <= q) ? acc[u][t][v] * scale : -__builtin_inff();
        }
      }
    }
  }
}

// ---------------------------------------------------------------------------
// 4) Row softmax. One wave per query row; writes P in bf16 and zero-fills up
//    to the 32-aligned causal tile boundary used by pv_k.
__global__ void softmax_k(const float* __restrict__ S, bf16* __restrict__ P) {
  const int lane = threadIdx.x & 31, w = threadIdx.x >> 5;
  const int rid = blockIdx.x * 8 + w;       // 0..8191 (b*s)
  const int q   = rid & (kS - 1);
  const float* srow = S + (size_t)rid * kS;
  bf16*        prow = P + (size_t)rid * kS;
  const int n = q + 1;                      // valid keys

  float m = -__builtin_inff();
  for (int k = lane; k < n; k += 32) m = fmaxf(m, srow[k]);
#pragma unroll
  for (int off = 16; off; off >>= 1) m = fmaxf(m, __shfl_xor(m, off, 32));

  float sum = 0.f;
  for (int k = lane; k < n; k += 32) sum += __expf(srow[k] - m);
#pragma unroll
  for (int off = 16; off; off >>= 1) sum += __shfl_xor(sum, off, 32);

  const float inv = 1.f / sum;
  const int fill = (((q & ~15) + 16) + 31) & ~31;   // == (q/32 + 1) * 32
  for (int k = lane; k < fill; k += 32)
    prow[k] = f2bf((k < n) ? __expf(srow[k] - m) * inv : 0.f);
}

// ---------------------------------------------------------------------------
// 5) O = P * V. One wave -> 32x64 tile; the 64x32 V tile (shared ebase) is
//    staged through LDS. k-loop runs to the block's max causal bound; waves
//    past their own bound skip the WMMAs (wave-uniform test, EXEC stays full).
__global__ void __launch_bounds__(256) pv_k(
    const bf16* __restrict__ P, const bf16* __restrict__ Vt,
    float* __restrict__ O) {
  __shared__ bf16 bufB[2][64 * kLP];
  const int tid = threadIdx.x;
  const int lane = tid & 31, w = tid >> 5;
  const int r = lane & 15, h = lane >> 4;
  const int qt    = blockIdx.x * 8 + w;   // 0..63 (32-row q tiles)
  const int ebase = blockIdx.y * 64;
  const int bz    = blockIdx.z;

  const bf16* prow0 = P + ((size_t)bz * kS + qt * 32 + r) * kS;
  const bf16* prow1 = prow0 + (size_t)16 * kS;

  // staging map: thread -> (col 0..63, kk 0,8,16,24), 8 elems (16B) each
  const int fcol = tid >> 2, fkk = (tid & 3) * 8;
  const bf16* vsrc = Vt + (size_t)bz * kD * kS + (size_t)(ebase + fcol) * kS + fkk;
  const int  foff = fcol * kLP + fkk;

  const int kmax_w   = (qt + 1) * 32;               // this wave's causal bound
  const int kmax_blk = (blockIdx.x * 8 + 8) * 32;   // block max (uniform)

  v8bf stg = *(const v8bf*)(vsrc);        // chunk kc=0
  *(v8bf*)(&bufB[0][foff]) = stg;
  __syncthreads();

  const v8f zero = {0.f, 0.f, 0.f, 0.f, 0.f, 0.f, 0.f, 0.f};
  v8f acc[2][4] = {{zero, zero, zero, zero}, {zero, zero, zero, zero}};

  int ib = 0;
  for (int kc = 0; kc < kmax_blk; kc += 32) {
    const bool more = (kc + 32) < kmax_blk;
    if (more) stg = *(const v8bf*)(vsrc + kc + 32);

    if (kc < kmax_w) {
      const v16bf a0 = load_a16(prow0, kc, h);
      const v16bf a1 = load_a16(prow1, kc, h);
#pragma unroll
      for (int t = 0; t < 4; ++t) {
        const v16bf b = load_b_lds(&bufB[ib][(t * 16 + r) * kLP], h);
        acc[0][t] = wmma_bf16(a0, b, acc[0][t]);
        acc[1][t] = wmma_bf16(a1, b, acc[1][t]);
      }
    }
    if (more) *(v8bf*)(&bufB[ib ^ 1][foff]) = stg;
    __syncthreads();
    ib ^= 1;
  }

#pragma unroll
  for (int u = 0; u < 2; ++u) {
#pragma unroll
    for (int t = 0; t < 4; ++t) {
      const int e = ebase + t * 16 + r;
#pragma unroll
      for (int v = 0; v < 8; ++v) {
        const int q = qt * 32 + u * 16 + v + 8 * h;
        O[((size_t)bz * kS + q) * kD + e] = acc[u][t][v];
      }
    }
  }
}

// ---------------------------------------------------------------------------
extern "C" void kernel_launch(void* const* d_in, const int* in_sizes, int n_in,
                              void* d_out, int out_size, void* d_ws, size_t ws_size,
                              hipStream_t stream) {
  const float* x  = (const float*)d_in[0];
  const float* Wq = (const float*)d_in[1];
  const float* Wk = (const float*)d_in[2];
  const float* Wv = (const float*)d_in[3];
  float* out = (float*)d_out;

  char* ws = (char*)d_ws;
  bf16*  Qb = (bf16*)(ws);
  bf16*  Kb = (bf16*)(ws + (size_t)16 * 1024 * 1024);
  bf16*  Vt = (bf16*)(ws + (size_t)32 * 1024 * 1024);
  bf16*  Wt = (bf16*)(ws + (size_t)48 * 1024 * 1024);
  float* S  = (float*)(ws + (size_t)56 * 1024 * 1024);
  bf16*  Xb = (bf16*)(ws + (size_t)56 * 1024 * 1024);  // overlaps S (dead early)
  bf16*  P  = (bf16*)(ws + (size_t)120 * 1024 * 1024);

  // 0) x -> bf16 (8M elements, 4/thread)
  xconv_k<<<dim3(8192), 256, 0, stream>>>(x, Xb);
  // 1) transpose + bf16-convert weights (3 x 1M elements)
  wtrans_k<<<dim3(4096, 3), 256, 0, stream>>>(Wq, Wk, Wv, Wt);
  // 2) Q/K/V projections: 256 m32-tiles x 16 n64-groups x 3 matrices
  qkv_k<<<dim3(32, 16, 3), 256, 0, stream>>>(Xb, Wt, Qb, Kb, Vt);
  // 3) causal scores: 64 q32-tiles x 8 k-groups(8 waves) x 4 batches
  scores_k<<<dim3(64, 8, 4), 256, 0, stream>>>(Qb, Kb, S);
  // 4) row softmax: 8192 rows, 8 rows (waves) per block
  softmax_k<<<dim3(1024), 256, 0, stream>>>(S, P);
  // 5) O = P.V: 8 q-superblocks x 16 e64-groups x 4 batches
  pv_k<<<dim3(8, 16, 4), 256, 0, stream>>>(P, Vt, out);
}
